// SRRNN_50946902065875
// MI455X (gfx1250) — matrix operand order repaired and checked by reading, not compiled, not run on hardware
//
#include <hip/hip_runtime.h>
#include <hip/hip_bf16.h>

// SRRNN: GRU cell + state-regularization (softmax over -||h-s||^2), bf16 WMMA.
// B=64, S=512, I=256, H=512, K=256, TEMP=1.0
// Single persistent kernel runs the whole 512-step scan: batch rows are
// independent, so each of the 4 blocks owns 16 rows and keeps h in LDS.
// x[:,s+1,:] is prefetched to LDS via global_load_async_to_lds_b128 (ASYNCcnt).

#define BDIM 64
#define SDIM 512
#define IDIM 256
#define HDIM 512
#define KST  256
#define TEMP_INV 1.0f
#define NTHREADS 512   // 16 waves

typedef __bf16 bf16;
typedef __attribute__((ext_vector_type(16))) __bf16 v16bf;
typedef __attribute__((ext_vector_type(8)))  __bf16 v8bf;
typedef __attribute__((ext_vector_type(8)))  float  v8f;

static __device__ __forceinline__ v8f wmma_bf16(v16bf a, v16bf b, v8f c) {
  // (neg_a, A, neg_b, B, c_mod, C, reuse_a, reuse_b)
  return __builtin_amdgcn_wmma_f32_16x16x32_bf16(false, a, false, b, (short)0, c,
                                                 false, false);
}

// A fragment (16x32 bf16, MxK): lane l holds row (l&15);
// element e -> k = k0 + (e/8)*16 + (l>>4)*8 + (e&7)  (two contiguous 16B runs).
static __device__ __forceinline__ v16bf load_fragA(const bf16* base, size_t ld, int k0) {
  const int lane = threadIdx.x & 31;
  const bf16* p = base + (size_t)(lane & 15) * ld + (size_t)k0 + ((lane >> 4) << 3);
  v8bf lo = *(const v8bf*)(p);
  v8bf hi = *(const v8bf*)(p + 16);
  v16bf r;
#pragma unroll
  for (int j = 0; j < 8; ++j) { r[j] = lo[j]; r[j + 8] = hi[j]; }
  return r;
}

// Same A fragment but sourced from f32 LDS (h kept canonical in f32), cvt on the fly.
static __device__ __forceinline__ v16bf load_fragA_f32(const float* base, size_t ld, int k0) {
  const int lane = threadIdx.x & 31;
  const float* p = base + (size_t)(lane & 15) * ld + (size_t)k0 + ((lane >> 4) << 3);
  v8f lo = *(const v8f*)(p);
  v8f hi = *(const v8f*)(p + 16);
  v16bf r;
#pragma unroll
  for (int j = 0; j < 8; ++j) { r[j] = (bf16)lo[j]; r[j + 8] = (bf16)hi[j]; }
  return r;
}

// B fragment (32x16 bf16, KxN), stored as "row n holds all k" (W rows / states rows).
// lane l holds column (l&15); element e -> k = k0 + (e/8)*16 + (l>>4)*8 + (e&7).
static __device__ __forceinline__ v16bf load_fragB(const bf16* base, size_t ld, int k0) {
  const int lane = threadIdx.x & 31;
  const bf16* p = base + (size_t)(lane & 15) * ld + (size_t)k0 + ((lane >> 4) << 3);
  v8bf lo = *(const v8bf*)(p);
  v8bf hi = *(const v8bf*)(p + 16);
  v16bf r;
#pragma unroll
  for (int j = 0; j < 8; ++j) { r[j] = lo[j]; r[j + 8] = hi[j]; }
  return r;
}

// Issue one 16B async global->LDS transfer per thread (512 threads = 16 rows x 512B).
static __device__ __forceinline__ void async_load_x_tile(const bf16* x_row_base,
                                                         bf16* dst_sh) {
  const int row = threadIdx.x >> 5;     // 0..15
  const int seg = threadIdx.x & 31;     // 0..31 (16B segments of a 512B row)
  const bf16* g = x_row_base + (size_t)row * (SDIM * IDIM) + seg * 8;
  // Generic shared pointer: low 32 bits are the wave-relative LDS byte offset.
  unsigned int lds = (unsigned int)(size_t)(dst_sh + row * IDIM + seg * 8);
  asm volatile("global_load_async_to_lds_b128 %0, %1, off"
               :: "v"(lds), "v"((unsigned long long)(size_t)g)
               : "memory");
}

// ---------------- preprocessing ----------------

__global__ void cvt_f32_bf16(const float* __restrict__ s, bf16* __restrict__ d, size_t n) {
  size_t i = (size_t)blockIdx.x * blockDim.x + threadIdx.x;
  const size_t st = (size_t)gridDim.x * blockDim.x;
  for (; i < n; i += st) d[i] = (bf16)s[i];
}

__global__ __launch_bounds__(128)
void states_prep(const float* __restrict__ st, bf16* __restrict__ st_bf,
                 bf16* __restrict__ stT_bf, float* __restrict__ st_sq) {
  const int k = blockIdx.x;
  __shared__ float red[128];
  float acc = 0.f;
  for (int h = threadIdx.x; h < HDIM; h += 128) {
    float v = st[(size_t)k * HDIM + h];
    st_bf[(size_t)k * HDIM + h] = (bf16)v;   // [K,H] for dist GEMM B
    stT_bf[(size_t)h * KST + k] = (bf16)v;   // [H,K] for projection GEMM B
    acc += v * v;
  }
  red[threadIdx.x] = acc;
  __syncthreads();
  for (int o = 64; o > 0; o >>= 1) {
    if (threadIdx.x < o) red[threadIdx.x] += red[threadIdx.x + o];
    __syncthreads();
  }
  if (threadIdx.x == 0) st_sq[k] = red[0];
}

// ---------------- persistent scan kernel ----------------

__global__ __launch_bounds__(NTHREADS)
void srrnn_scan(const bf16* __restrict__ x_bf,   // [B,S,I]
                const bf16* __restrict__ wih_bf, // [3H,I]
                const bf16* __restrict__ whh_bf, // [3H,H]
                const bf16* __restrict__ st_bf,  // [K,H]
                const bf16* __restrict__ stT_bf, // [H,K]
                const float* __restrict__ st_sq, // [K]
                const float* __restrict__ h0,    // [B,H]
                const float* __restrict__ b_ih,
                const float* __restrict__ b_hh,
                float* __restrict__ out,         // [B,S,H]
                float* __restrict__ probs,       // [B,S,K]
                float* __restrict__ h_n) {       // [B,H]
  __shared__ __align__(16) bf16  xs[2][16 * IDIM];     // 16 KB: double-buffered x slice
  __shared__ __align__(32) float h_f[16 * HDIM];       // 32 KB: canonical h (f32)
  __shared__ __align__(16) bf16  hc_bf[16 * HDIM];     // 16 KB: h_cand (bf16)
  __shared__ __align__(32) float e_lds[16 * KST];      // 16 KB: logits
  __shared__ __align__(16) bf16  p_lds[16 * KST];      //  8 KB: probs (bf16)

  const int m0 = blockIdx.x * 16;          // this block's batch rows
  const int wave = threadIdx.x >> 5;       // 0..15
  const int lane = threadIdx.x & 31;
  const int tid = threadIdx.x;

  // h <- h_0 (rows m0..m0+15 are contiguous)
  for (int i = tid; i < 16 * HDIM; i += NTHREADS)
    h_f[i] = h0[(size_t)m0 * HDIM + i];
  // async preload x[:, 0, :]
  async_load_x_tile(x_bf + (size_t)m0 * SDIM * IDIM, &xs[0][0]);

  for (int s = 0; s < SDIM; ++s) {
    const int cur = s & 1;
    asm volatile("s_wait_asynccnt 0x0" ::: "memory");   // my x transfers landed
    __syncthreads();                                    // everyone's x + h ready

    if (s + 1 < SDIM)  // prefetch next x slice into the other buffer
      async_load_x_tile(x_bf + ((size_t)m0 * SDIM + (s + 1)) * IDIM, &xs[cur ^ 1][0]);

    // ---- Phase G: gates -> h_cand (2 hidden tiles per wave) ----
    for (int t = wave; t < HDIM / 16; t += 16) {
      const int n0 = t * 16;
      v8f ar = {}, az = {}, agi = {}, agh = {};
      const bf16* wr = wih_bf + (size_t)(0 * HDIM + n0) * IDIM;
      const bf16* wz = wih_bf + (size_t)(1 * HDIM + n0) * IDIM;
      const bf16* wn = wih_bf + (size_t)(2 * HDIM + n0) * IDIM;
      for (int k0 = 0; k0 < IDIM; k0 += 32) {
        v16bf a = load_fragA(&xs[cur][0], IDIM, k0);
        ar  = wmma_bf16(a, load_fragB(wr, IDIM, k0), ar);
        az  = wmma_bf16(a, load_fragB(wz, IDIM, k0), az);
        agi = wmma_bf16(a, load_fragB(wn, IDIM, k0), agi);
      }
      const bf16* ur = whh_bf + (size_t)(0 * HDIM + n0) * HDIM;
      const bf16* uz = whh_bf + (size_t)(1 * HDIM + n0) * HDIM;
      const bf16* un = whh_bf + (size_t)(2 * HDIM + n0) * HDIM;
      for (int k0 = 0; k0 < HDIM; k0 += 32) {
        __builtin_prefetch(ur + k0 + 64, 0, 1);   // global_prefetch_b8 on L2-resident weights
        v16bf a = load_fragA_f32(h_f, HDIM, k0);
        ar  = wmma_bf16(a, load_fragB(ur, HDIM, k0), ar);
        az  = wmma_bf16(a, load_fragB(uz, HDIM, k0), az);
        agh = wmma_bf16(a, load_fragB(un, HDIM, k0), agh);
      }
      // C/D layout: lane l, vreg v -> row m = v + 8*(l/16), col n = l&15
      const int n = n0 + (lane & 15);
      const float br   = b_ih[n] + b_hh[n];
      const float bz   = b_ih[HDIM + n] + b_hh[HDIM + n];
      const float bin_ = b_ih[2 * HDIM + n];
      const float bhn  = b_hh[2 * HDIM + n];
#pragma unroll
      for (int v = 0; v < 8; ++v) {
        const int m = v + ((lane >> 4) << 3);
        float r  = 1.f / (1.f + __expf(-(ar[v] + br)));
        float z  = 1.f / (1.f + __expf(-(az[v] + bz)));
        float nn = tanhf(agi[v] + bin_ + r * (agh[v] + bhn));
        float hc = (1.f - z) * nn + z * h_f[m * HDIM + n];
        hc_bf[m * HDIM + n] = (bf16)hc;
      }
    }
    __syncthreads();

    // ---- Phase D: logits E = (2*hc.s - |s|^2)/TEMP, one K-tile per wave ----
    {
      const int t = wave;
      v8f acc = {};
      const bf16* sb = st_bf + (size_t)(t * 16) * HDIM;
      for (int k0 = 0; k0 < HDIM; k0 += 32)
        acc = wmma_bf16(load_fragA(hc_bf, HDIM, k0), load_fragB(sb, HDIM, k0), acc);
      const int n = t * 16 + (lane & 15);
      const float ss = st_sq[n];
#pragma unroll
      for (int v = 0; v < 8; ++v) {
        const int m = v + ((lane >> 4) << 3);
        e_lds[m * KST + n] = (2.f * acc[v] - ss) * TEMP_INV;
      }
    }
    __syncthreads();

    // ---- Phase S: row softmax (one row per wave), write probs + bf16 P ----
    {
      const int r = wave;
      float mx = -3.4e38f;
      for (int c = lane; c < KST; c += 32) mx = fmaxf(mx, e_lds[r * KST + c]);
      for (int o = 16; o > 0; o >>= 1) mx = fmaxf(mx, __shfl_xor(mx, o, 32));
      float sum = 0.f;
      for (int c = lane; c < KST; c += 32) {
        float ex = __expf(e_lds[r * KST + c] - mx);
        e_lds[r * KST + c] = ex;
        sum += ex;
      }
      for (int o = 16; o > 0; o >>= 1) sum += __shfl_xor(sum, o, 32);
      const float inv = 1.f / sum;
      const size_t pb = ((size_t)(m0 + r) * SDIM + s) * KST;
      for (int c = lane; c < KST; c += 32) {
        float p = e_lds[r * KST + c] * inv;
        p_lds[r * KST + c] = (bf16)p;
        probs[pb + c] = p;
      }
    }
    __syncthreads();

    // ---- Phase H: h_new = P @ states (2 hidden tiles per wave) ----
    for (int t = wave; t < HDIM / 16; t += 16) {
      v8f acc = {};
      const bf16* sb = stT_bf + (size_t)(t * 16) * KST;
      for (int k0 = 0; k0 < KST; k0 += 32)
        acc = wmma_bf16(load_fragA(p_lds, KST, k0), load_fragB(sb, KST, k0), acc);
      const int n = t * 16 + (lane & 15);
#pragma unroll
      for (int v = 0; v < 8; ++v) {
        const int m = v + ((lane >> 4) << 3);
        const float hv = acc[v];
        out[((size_t)(m0 + m) * SDIM + s) * HDIM + n] = hv;
        h_f[m * HDIM + n] = hv;   // consumed next step after loop-top barrier
      }
    }
  }

  __syncthreads();
  for (int i = tid; i < 16 * HDIM; i += NTHREADS)
    h_n[(size_t)m0 * HDIM + i] = h_f[i];
}

// ---------------- launch ----------------

extern "C" void kernel_launch(void* const* d_in, const int* in_sizes, int n_in,
                              void* d_out, int out_size, void* d_ws, size_t ws_size,
                              hipStream_t stream) {
  (void)in_sizes; (void)n_in; (void)out_size; (void)ws_size;
  const float* input_ = (const float*)d_in[0];   // [B,S,I]
  const float* h0     = (const float*)d_in[1];   // [B,H]
  const float* wih    = (const float*)d_in[2];   // [3H,I]
  const float* whh    = (const float*)d_in[3];   // [3H,H]
  const float* bih    = (const float*)d_in[4];   // [3H]
  const float* bhh    = (const float*)d_in[5];   // [3H]
  const float* states = (const float*)d_in[6];   // [K,H]

  float* out   = (float*)d_out;                          // [B,S,H]
  float* h_n   = out + (size_t)BDIM * SDIM * HDIM;       // [B,H]
  float* probs = h_n + (size_t)BDIM * HDIM;              // [B,S,K]

  char* w = (char*)d_ws;
  auto alloc = [&](size_t bytes) -> char* {
    char* p = w;
    w += (bytes + 255) & ~(size_t)255;
    return p;
  };
  bf16*  x_bf   = (bf16*)alloc(sizeof(bf16) * (size_t)BDIM * SDIM * IDIM);
  bf16*  wih_bf = (bf16*)alloc(sizeof(bf16) * (size_t)3 * HDIM * IDIM);
  bf16*  whh_bf = (bf16*)alloc(sizeof(bf16) * (size_t)3 * HDIM * HDIM);
  bf16*  st_bf  = (bf16*)alloc(sizeof(bf16) * (size_t)KST * HDIM);
  bf16*  stT_bf = (bf16*)alloc(sizeof(bf16) * (size_t)HDIM * KST);
  float* st_sq  = (float*)alloc(sizeof(float) * KST);

  cvt_f32_bf16<<<4096, 256, 0, stream>>>(input_, x_bf, (size_t)BDIM * SDIM * IDIM);
  cvt_f32_bf16<<<512, 256, 0, stream>>>(wih, wih_bf, (size_t)3 * HDIM * IDIM);
  cvt_f32_bf16<<<512, 256, 0, stream>>>(whh, whh_bf, (size_t)3 * HDIM * HDIM);
  states_prep<<<KST, 128, 0, stream>>>(states, st_bf, stT_bf, st_sq);

  // One persistent kernel: 4 blocks x 512 threads run the whole 512-step scan.
  srrnn_scan<<<BDIM / 16, NTHREADS, 0, stream>>>(x_bf, wih_bf, whh_bf, st_bf, stT_bf,
                                                 st_sq, h0, bih, bhh, out, probs, h_n);
}
